// MOE_21036749816504
// MI455X (gfx1250) — matrix operand order repaired and checked by reading, not compiled, not run on hardware
//
#include <hip/hip_runtime.h>

#define N_TOK 4096
#define DIM_D 1024
#define DIM_H 4096
#define DIM_E 8
#define DIM_O 1024
#define TOPK  2

#define BM 128
#define BN 128
#define KT 32
#define LDSS 40   // halves per LDS row: 32 data + 8 pad -> 80B rows (16B aligned)

#define SLOT_MAX (N_TOK * TOPK + DIM_E * BM)   // padded per-expert regions

#if __has_builtin(__builtin_amdgcn_tensor_load_to_lds)
#define HAS_TDM 1
#else
#define HAS_TDM 0
#endif

typedef __attribute__((ext_vector_type(16))) _Float16     v16h;
typedef __attribute__((ext_vector_type(8)))  float        v8f;
typedef __attribute__((ext_vector_type(4)))  unsigned int u32x4;
typedef __attribute__((ext_vector_type(8)))  int          i32x8;
typedef __attribute__((ext_vector_type(4)))  int          i32x4;

union FragH { v16h v; uint4 u[2]; };

// ---------------------------------------------------------------------------
// Router: one wave (32 lanes) per token. logits = x @ Wg + bg, softmax, top-2.
// ---------------------------------------------------------------------------
__global__ __launch_bounds__(256)
void router_kernel(const float* __restrict__ x, const float* __restrict__ Wg,
                   const float* __restrict__ bg,
                   int* __restrict__ topi, float* __restrict__ topw,
                   int* __restrict__ counts) {
    const int wave = threadIdx.x >> 5;
    const int lane = threadIdx.x & 31;
    const int n    = blockIdx.x * 8 + wave;

    float acc[DIM_E];
#pragma unroll
    for (int e = 0; e < DIM_E; ++e) acc[e] = 0.f;

    for (int d = lane; d < DIM_D; d += 32) {
        const float xv = x[(size_t)n * DIM_D + d];
#pragma unroll
        for (int e = 0; e < DIM_E; ++e)
            acc[e] += xv * Wg[(size_t)d * DIM_E + e];
    }
#pragma unroll
    for (int e = 0; e < DIM_E; ++e)
        for (int off = 16; off > 0; off >>= 1)
            acc[e] += __shfl_xor(acc[e], off, 32);

    if (lane == 0) {
        float m = -1e30f;
#pragma unroll
        for (int e = 0; e < DIM_E; ++e) { acc[e] += bg[e]; m = fmaxf(m, acc[e]); }
        float g[DIM_E]; float s = 0.f;
#pragma unroll
        for (int e = 0; e < DIM_E; ++e) { g[e] = __expf(acc[e] - m); s += g[e]; }
        const float inv = 1.f / s;
        int i0 = 0;
#pragma unroll
        for (int e = 1; e < DIM_E; ++e) if (g[e] > g[i0]) i0 = e;
        int i1 = (i0 == 0) ? 1 : 0;
#pragma unroll
        for (int e = 0; e < DIM_E; ++e) if (e != i0 && g[e] > g[i1]) i1 = e;
        topi[n * 2 + 0] = i0;  topw[n * 2 + 0] = g[i0] * inv;
        topi[n * 2 + 1] = i1;  topw[n * 2 + 1] = g[i1] * inv;
        atomicAdd(&counts[i0], 1);
        atomicAdd(&counts[i1], 1);
    }
}

// Tile-padded exclusive scan over 8 expert counts.
__global__ void scan_kernel(const int* __restrict__ counts, int* __restrict__ baseo) {
    if (threadIdx.x == 0) {
        int run = 0;
        for (int e = 0; e < DIM_E; ++e) {
            baseo[e] = run;
            run += ((counts[e] + BM - 1) / BM) * BM;
        }
    }
}

// Assign each (token, k) a slot inside its expert's padded region.
__global__ __launch_bounds__(256)
void build_kernel(const int* __restrict__ topi, const int* __restrict__ baseo,
                  int* __restrict__ counts2,
                  int* __restrict__ slot_token, int* __restrict__ slot_of) {
    const int idx = blockIdx.x * blockDim.x + threadIdx.x;
    if (idx >= N_TOK * TOPK) return;
    const int e   = topi[idx];
    const int pos = baseo[e] + atomicAdd(&counts2[e], 1);
    slot_token[pos] = idx >> 1;
    slot_of[idx]    = pos;
}

// f32 -> f16 linear conversion (x only).
__global__ __launch_bounds__(256)
void cast_kernel(const float* __restrict__ src, _Float16* __restrict__ dst, long n) {
    const long i = ((long)blockIdx.x * blockDim.x + threadIdx.x) * 4;
    if (i >= n) return;
    const float4 v = *(const float4*)(src + i);
    dst[i + 0] = (_Float16)v.x;
    dst[i + 1] = (_Float16)v.y;
    dst[i + 2] = (_Float16)v.z;
    dst[i + 3] = (_Float16)v.w;
}

// Fused f32->f16 transpose: src [E][R][C] f32 -> dst [E][C][R] f16 (32x32 tiles).
template <int R, int C>
__global__ __launch_bounds__(256)
void transpose_cast_kernel(const float* __restrict__ src, _Float16* __restrict__ dst) {
    __shared__ float tile[32][33];
    const int e  = blockIdx.z;
    const int c0 = blockIdx.x * 32;
    const int r0 = blockIdx.y * 32;
    const int tx = threadIdx.x & 31;
    const int ty = threadIdx.x >> 5;   // 0..7
#pragma unroll
    for (int i = 0; i < 4; ++i) {
        const int r = ty + i * 8;
        tile[r][tx] = src[((size_t)e * R + (r0 + r)) * C + (c0 + tx)];
    }
    __syncthreads();
#pragma unroll
    for (int i = 0; i < 4; ++i) {
        const int c = ty + i * 8;
        dst[((size_t)e * C + (c0 + c)) * R + (r0 + tx)] = (_Float16)tile[tx][c];
    }
}

// ---------------------------------------------------------------------------
// Tiled WMMA GEMM over one expert's compacted token slots.
//   C[row, n] = sum_k A[row, k] * Bt[e, n, k]   (Bt is pre-transposed weights)
// Block: 256 threads = 8 waves; tile BM=128 x BN=128, K-step 32.
// Wave (wr 0..3, wc 0..1) computes a 32x64 sub-tile = 2x4 WMMA 16x16 tiles
// (8 WMMAs per K-step from 2 A-frags + 4 B-frags; frag reuse 4x/2x).
// Double-buffered staging: wave 0 issues the next B tile via the Tensor Data
// Mover (LDS padding reproduces the 80B row stride); all threads prefetch the
// next (gathered) A rows into registers; s_wait_tensorcnt + one barrier after
// compute overlaps the DMA with the WMMA block.
// ---------------------------------------------------------------------------
template <int KTOT, int NCOLS, bool GATHER, bool FUSE_RELU_F16>
__global__ __launch_bounds__(256)
void moe_gemm_kernel(const _Float16* __restrict__ A,
                     const int* __restrict__ slot_token,
                     const _Float16* __restrict__ Bt,   // [E][NCOLS][KTOT]
                     const float* __restrict__ bias,
                     const int* __restrict__ counts,
                     const int* __restrict__ baseo,
                     void* __restrict__ outp) {
    const int e   = blockIdx.z;
    const int cnt = counts[e];
    const int rowTileBase = blockIdx.y * BM;
    if (rowTileBase >= cnt) return;          // over-launched tile row: skip
    const int n0       = blockIdx.x * BN;
    const int baseSlot = baseo[e];

    __shared__ _Float16 sA[2][BM * LDSS] __attribute__((aligned(16)));
    __shared__ _Float16 sB[2][BN * LDSS] __attribute__((aligned(16)));

    const int tid  = threadIdx.x;
    const int lane = tid & 31;
    const int wave = tid >> 5;
    const int wr   = wave >> 2;        // 0..1  -> not used; see below
    const int wrM  = wave >> 1;        // 0..3  -> 32-row band
    const int wcN  = wave & 1;         // 0..1  -> 64-col band
    (void)wr;

    v8f acc[2][4] = {};

    // A staging: 2 (row, 16B-chunk) pairs per thread covering 128 rows x 64B.
    int aOff[2]; bool aValid[2]; const _Float16* aPtr[2];
#pragma unroll
    for (int s = 0; s < 2; ++s) {
        const int idx  = tid + s * 256;
        const int arow = idx >> 2;
        const int achk = idx & 3;
        aOff[s]   = arow * LDSS + achk * 8;
        aValid[s] = (rowTileBase + arow) < cnt;
        aPtr[s]   = A;
        if (aValid[s]) {
            long rowIdx;
            if (GATHER) rowIdx = slot_token[baseSlot + rowTileBase + arow];
            else        rowIdx = baseSlot + rowTileBase + arow;
            aPtr[s] = A + (size_t)rowIdx * KTOT + achk * 8;
        }
    }

#if HAS_TDM
    const unsigned ldsBaddr[2] = { (unsigned)(size_t)(void*)&sB[0][0],
                                   (unsigned)(size_t)(void*)&sB[1][0] };
    const unsigned long long gBbase =
        (unsigned long long)(size_t)(Bt + ((size_t)e * NCOLS + n0) * KTOT);
    // D# group 1 (constant): data_size=2B, pad every 16 DW by 4 DW
    // (row 64B + 16B pad = 80B LDS stride), tensor_dim0=KTOT,
    // dim0_stride=KTOT, tensor_dim1=NCOLS, tile_dim0=KT, tile_dim1=BN.
    i32x8 g1;
    g1[0] = (int)((1u << 16) | (1u << 20) | (3u << 22) | (3u << 25));
    g1[1] = (int)(((unsigned)KTOT & 0xFFFFu) << 16);
    g1[2] = (int)((((unsigned)KTOT >> 16) & 0xFFFFu) |
                  (((unsigned)NCOLS & 0xFFFFu) << 16));
    g1[3] = (int)((((unsigned)NCOLS >> 16) & 0xFFFFu) | ((unsigned)KT << 16));
    g1[4] = BN;          // tile_dim1 = 128, tile_dim2 = 0
    g1[5] = KTOT;        // tensor_dim0_stride[31:0]
    g1[6] = 0;
    g1[7] = 0;
    const i32x4 gz4 = {0, 0, 0, 0};
    const i32x8 gz8 = {0, 0, 0, 0, 0, 0, 0, 0};
    auto issueTDM = [&](int kk, int buf) {
        const unsigned long long ga = gBbase + (unsigned long long)kk * 2ull;
        u32x4 g0;
        g0[0] = 1u;                                   // count=1
        g0[1] = ldsBaddr[buf];                        // lds_addr
        g0[2] = (unsigned)ga;                         // global_addr lo
        g0[3] = (unsigned)(ga >> 32) | (2u << 30);    // global_addr hi | type=2
        __builtin_amdgcn_tensor_load_to_lds(g0, g1, gz4, gz4, gz8, 0);
    };
#else
    int bOff[2]; const _Float16* bPtr[2];
#pragma unroll
    for (int s = 0; s < 2; ++s) {
        const int idx  = tid + s * 256;
        const int brow = idx >> 2;
        const int bchk = idx & 3;
        bOff[s] = brow * LDSS + bchk * 8;
        bPtr[s] = Bt + ((size_t)e * NCOLS + n0 + brow) * KTOT + bchk * 8;
    }
#endif

    // ---- prologue: stage tile 0 into buffer 0 -----------------------------
    {
#if HAS_TDM
        if (tid < 32) issueTDM(0, 0);
#else
#pragma unroll
        for (int s = 0; s < 2; ++s)
            *(uint4*)&sB[0][bOff[s]] = *(const uint4*)bPtr[s];
#endif
#pragma unroll
        for (int s = 0; s < 2; ++s) {
            uint4 av = make_uint4(0u, 0u, 0u, 0u);
            if (aValid[s]) av = *(const uint4*)aPtr[s];
            *(uint4*)&sA[0][aOff[s]] = av;
        }
#if HAS_TDM
        if (tid < 32) __builtin_amdgcn_s_wait_tensorcnt(0);
#endif
        __syncthreads();
    }

    // ---- main pipelined K loop -------------------------------------------
    const int laneR = lane & 15;
    const int laneH = lane >> 4;

    for (int k0 = 0; k0 < KTOT; k0 += KT) {
        const int  cur     = (k0 / KT) & 1;
        const int  nxt     = cur ^ 1;
        const bool hasNext = (k0 + KT) < KTOT;

        // Prefetch next tiles (DMA for B, registers for A) before compute.
        uint4 avn[2] = { make_uint4(0u,0u,0u,0u), make_uint4(0u,0u,0u,0u) };
#if !HAS_TDM
        uint4 bvn[2] = { make_uint4(0u,0u,0u,0u), make_uint4(0u,0u,0u,0u) };
#endif
        if (hasNext) {
#if HAS_TDM
            if (tid < 32) issueTDM(k0 + KT, nxt);
#else
#pragma unroll
            for (int s = 0; s < 2; ++s) bvn[s] = *(const uint4*)(bPtr[s] + k0 + KT);
#endif
#pragma unroll
            for (int s = 0; s < 2; ++s)
                if (aValid[s]) avn[s] = *(const uint4*)(aPtr[s] + k0 + KT);
        }

        // Compute on current buffers: 2 A-frags, 4 B-frags, 8 WMMAs.
        FragH fa[2];
#pragma unroll
        for (int mi = 0; mi < 2; ++mi) {
            const int row = wrM * 32 + mi * 16 + laneR;
            const int kb  = laneH * 8;
            fa[mi].u[0] = *(const uint4*)&sA[cur][row * LDSS + kb];
            fa[mi].u[1] = *(const uint4*)&sA[cur][row * LDSS + kb + 16];
        }
#pragma unroll
        for (int ni = 0; ni < 4; ++ni) {
            FragH fb;
            const int nc  = wcN * 64 + ni * 16 + laneR;
            const int kb2 = laneH * 8;
            fb.u[0] = *(const uint4*)&sB[cur][nc * LDSS + kb2];
            fb.u[1] = *(const uint4*)&sB[cur][nc * LDSS + kb2 + 16];
            acc[0][ni] = __builtin_amdgcn_wmma_f32_16x16x32_f16(
                false, fa[0].v, false, fb.v, (short)0, acc[0][ni], false, false);
            acc[1][ni] = __builtin_amdgcn_wmma_f32_16x16x32_f16(
                false, fa[1].v, false, fb.v, (short)0, acc[1][ni], false, false);
        }

        // Publish next tiles; DMA + global loads overlapped the WMMAs above.
        if (hasNext) {
#if !HAS_TDM
#pragma unroll
            for (int s = 0; s < 2; ++s) *(uint4*)&sB[nxt][bOff[s]] = bvn[s];
#endif
#pragma unroll
            for (int s = 0; s < 2; ++s) *(uint4*)&sA[nxt][aOff[s]] = avn[s];
        }
#if HAS_TDM
        if (tid < 32) __builtin_amdgcn_s_wait_tensorcnt(0);
#endif
        __syncthreads();
    }

    // Epilogue. C/D layout: VGPR r -> M = r + 8*(lane>=16), N = lane&15.
    const int mBase = rowTileBase + wrM * 32 + laneH * 8;
    float bv[4];
#pragma unroll
    for (int ni = 0; ni < 4; ++ni)
        bv[ni] = bias[(size_t)e * NCOLS + n0 + wcN * 64 + ni * 16 + laneR];

    if (FUSE_RELU_F16) {
        _Float16* outH = (_Float16*)outp;
#pragma unroll
        for (int mi = 0; mi < 2; ++mi)
#pragma unroll
            for (int r = 0; r < 8; ++r) {
                const int rowT = mBase + mi * 16 + r;
                if (rowT < cnt) {
                    const size_t base = (size_t)(baseSlot + rowT) * NCOLS;
#pragma unroll
                    for (int ni = 0; ni < 4; ++ni) {
                        float v = acc[mi][ni][r] + bv[ni];
                        v = v > 0.f ? v : 0.f;
                        outH[base + n0 + wcN * 64 + ni * 16 + laneR] = (_Float16)v;
                    }
                }
            }
    } else {
        float* outY = (float*)outp;
#pragma unroll
        for (int mi = 0; mi < 2; ++mi)
#pragma unroll
            for (int r = 0; r < 8; ++r) {
                const int rowT = mBase + mi * 16 + r;
                if (rowT < cnt) {
                    const size_t base = (size_t)(baseSlot + rowT) * NCOLS;
#pragma unroll
                    for (int ni = 0; ni < 4; ++ni)
                        outY[base + n0 + wcN * 64 + ni * 16 + laneR] =
                            acc[mi][ni][r] + bv[ni];
                }
            }
    }
}

// out[n, o] = w0 * y[slot0, o] + w1 * y[slot1, o]   (deterministic combine)
__global__ __launch_bounds__(256)
void combine_kernel(const float* __restrict__ y, const int* __restrict__ slot_of,
                    const float* __restrict__ topw, float* __restrict__ out) {
    const long idx = (long)blockIdx.x * blockDim.x + threadIdx.x;
    if (idx >= (long)N_TOK * DIM_O) return;
    const int n = (int)(idx / DIM_O);
    const int o = (int)(idx % DIM_O);
    const int s0 = slot_of[n * 2 + 0];
    const int s1 = slot_of[n * 2 + 1];
    out[idx] = topw[n * 2 + 0] * y[(size_t)s0 * DIM_O + o]
             + topw[n * 2 + 1] * y[(size_t)s1 * DIM_O + o];
}

// ---------------------------------------------------------------------------
extern "C" void kernel_launch(void* const* d_in, const int* in_sizes, int n_in,
                              void* d_out, int out_size, void* d_ws, size_t ws_size,
                              hipStream_t stream) {
    const float* x  = (const float*)d_in[0];
    const float* Wg = (const float*)d_in[1];
    const float* bg = (const float*)d_in[2];
    const float* W1 = (const float*)d_in[3];
    const float* b1 = (const float*)d_in[4];
    const float* W2 = (const float*)d_in[5];
    const float* b2 = (const float*)d_in[6];
    float* out = (float*)d_out;

    char* ws = (char*)d_ws;
    auto take = [&](size_t bytes) -> char* {
        char* p = ws;
        ws += (bytes + 255) & ~(size_t)255;
        return p;
    };
    int*       counts     = (int*)take(DIM_E * sizeof(int));
    int*       counts2    = (int*)take(DIM_E * sizeof(int));
    int*       baseo      = (int*)take(DIM_E * sizeof(int));
    int*       topi       = (int*)take((size_t)N_TOK * TOPK * sizeof(int));
    float*     topw       = (float*)take((size_t)N_TOK * TOPK * sizeof(float));
    int*       slot_token = (int*)take((size_t)SLOT_MAX * sizeof(int));
    int*       slot_of    = (int*)take((size_t)N_TOK * TOPK * sizeof(int));
    _Float16*  xb         = (_Float16*)take((size_t)N_TOK * DIM_D * sizeof(_Float16));
    _Float16*  W1t        = (_Float16*)take((size_t)DIM_E * DIM_D * DIM_H * sizeof(_Float16));
    _Float16*  W2t        = (_Float16*)take((size_t)DIM_E * DIM_H * DIM_O * sizeof(_Float16));
    _Float16*  hidden     = (_Float16*)take((size_t)SLOT_MAX * DIM_H * sizeof(_Float16));
    float*     ybuf       = (float*)take((size_t)SLOT_MAX * DIM_O * sizeof(float));

    (void)hipMemsetAsync(counts,  0, DIM_E * sizeof(int), stream);
    (void)hipMemsetAsync(counts2, 0, DIM_E * sizeof(int), stream);

    // Precision cast of x; fused cast+transpose of the expert weights.
    {
        const long nx = (long)N_TOK * DIM_D;
        cast_kernel<<<dim3((unsigned)((nx / 4 + 255) / 256)), 256, 0, stream>>>(x, xb, nx);
    }
    transpose_cast_kernel<DIM_D, DIM_H>
        <<<dim3(DIM_H / 32, DIM_D / 32, DIM_E), 256, 0, stream>>>(W1, W1t);
    transpose_cast_kernel<DIM_H, DIM_O>
        <<<dim3(DIM_O / 32, DIM_H / 32, DIM_E), 256, 0, stream>>>(W2, W2t);

    router_kernel<<<dim3(N_TOK / 8), 256, 0, stream>>>(x, Wg, bg, topi, topw, counts);
    scan_kernel<<<dim3(1), 32, 0, stream>>>(counts, baseo);
    build_kernel<<<dim3((N_TOK * TOPK + 255) / 256), 256, 0, stream>>>(
        topi, baseo, counts2, slot_token, slot_of);

    // GEMM1: hidden = relu(gather(xb) @ W1 + b1), f16 out.
    moe_gemm_kernel<DIM_D, DIM_H, true, true>
        <<<dim3(DIM_H / BN, N_TOK / BM, DIM_E), 256, 0, stream>>>(
            xb, slot_token, W1t, b1, counts, baseo, (void*)hidden);

    // GEMM2: y = hidden @ W2 + b2, f32 out.
    moe_gemm_kernel<DIM_H, DIM_O, false, false>
        <<<dim3(DIM_O / BN, N_TOK / BM, DIM_E), 256, 0, stream>>>(
            hidden, nullptr, W2t, b2, counts, baseo, (void*)ybuf);

    combine_kernel<<<dim3((unsigned)(((long)N_TOK * DIM_O + 255) / 256)), 256, 0, stream>>>(
        ybuf, slot_of, topw, out);
}
